// WALinear_66778151518366
// MI455X (gfx1250) — matrix operand order repaired and checked by reading, not compiled, not run on hardware
//
#include <hip/hip_runtime.h>

typedef __attribute__((ext_vector_type(16))) _Float16 v16h;
typedef __attribute__((ext_vector_type(8)))  _Float16 v8h;
typedef __attribute__((ext_vector_type(2)))  _Float16 v2h;
typedef __attribute__((ext_vector_type(8)))  float    v8f;

// Problem constants (from reference setup_inputs)
constexpr int I_DIM = 4096;                 // in_features (K)
constexpr int O_DIM = 11008;                // out_features (N)
constexpr int M_TOT = 4 * 2048;             // B*S (M)
constexpr int GROUP = 64;                   // quant group size
constexpr int ZS_W  = 2 * I_DIM / GROUP;    // zeros_scales row width = 128

// Tiling
constexpr int BM = 128;
constexpr int BN = 128;
constexpr int BK = 64;                      // == GROUP
constexpr int LDT = BK + 16;                // LDS row stride in halves (160B, 16B aligned)

__global__ __launch_bounds__(256, 2)
void w4a16_wmma_kernel(const _Float16* __restrict__ x,
                       const int*      __restrict__ wq,
                       const _Float16* __restrict__ zs,
                       const _Float16* __restrict__ bias,
                       _Float16*       __restrict__ y)
{
    // Double-buffered tiles: 2 * (128*80 + 128*80) * 2B = 80 KB LDS
    __shared__ _Float16 As[2][BM * LDT];    // x tile (f16)
    __shared__ _Float16 Bs[2][BN * LDT];    // dequantized W tile (f16)

    const int tid    = threadIdx.x;
    const int lane   = tid & 31;
    const int wave   = tid >> 5;            // 0..7
    const int wm     = wave >> 2;           // 0..1 (M dir)
    const int wn     = wave & 3;            // 0..3 (N dir)
    const int lane16 = lane & 15;
    const int sel    = lane >> 4;

    const int bn0 = blockIdx.x * BN;
    const int bm0 = blockIdx.y * BM;

    v8f acc[4][2];
    const v8f vzero = {0.f, 0.f, 0.f, 0.f, 0.f, 0.f, 0.f, 0.f};
#pragma unroll
    for (int i = 0; i < 4; ++i)
#pragma unroll
        for (int j = 0; j < 2; ++j)
            acc[i][j] = vzero;

    // ---- per-thread W-dequant assignment: 2 threads per row, 16 int32 each ----
    const int wr   = tid >> 1;              // row in tile 0..127
    const int wh   = tid & 1;               // which 16-int half of the row
    const int orow = bn0 + wr;
    const _Float16* zrow_base = zs + (size_t)orow * ZS_W;
    const int*      wrow_base = wq + (size_t)orow * (I_DIM / 2) + wh * 16;

    // Issue async x-tile copy (global -> LDS, ASYNCcnt-tracked, no VGPR staging)
    auto issue_x = [&](int buf, int kb) {
#pragma unroll
        for (int it = 0; it < 4; ++it) {
            int c   = tid + it * 256;       // 1024 chunks of 8 halves
            int r   = c >> 3;
            int col = (c & 7) * 8;
            unsigned lds_addr = (unsigned)(unsigned long long)(&As[buf][r * LDT + col]);
            const void* gp = (const void*)(x + (size_t)(bm0 + r) * I_DIM + kb + col);
            asm volatile("global_load_async_to_lds_b128 %0, %1, off"
                         :: "v"(lds_addr), "v"(gp) : "memory");
        }
    };

    // Load packed weights + (zero,scale) dword for a K-step into registers
    int4     wl[4];
    unsigned zsv;
    auto load_w = [&](int kb) {
        const int* wrow = wrow_base + (kb >> 1);
#pragma unroll
        for (int q4 = 0; q4 < 4; ++q4)
            wl[q4] = *(const int4*)(wrow + q4 * 4);
        zsv = *(const unsigned*)(zrow_base + 2 * (kb / GROUP));  // {zero, scale}
    };

    // Dequantize registers -> LDS using the 0x6400 fp16 bias trick:
    //   bitcast(0x6400 | q) == 1024 + q  (exact), so
    //   (q - z)*s == h * s + c,  c = -(1024 + z)*s   (one v_pk_fma_f16 per int32)
    auto dequant_store = [&](int buf) {
        union { unsigned u; v2h h; } uz, us;
        uz.u = zsv & 0xFFFFu;                 // zero (f16 in low half)
        us.u = zsv >> 16;                     // scale
        float zf = (float)uz.h.x;
        float sf = (float)us.h.x;
        _Float16 ch = (_Float16)(-(1024.0f + zf) * sf);
        _Float16 sh = (_Float16)sf;
        v2h s2 = {sh, sh};
        v2h c2 = {ch, ch};
#pragma unroll
        for (int q4 = 0; q4 < 4; ++q4) {
            int vals[4] = {wl[q4].x, wl[q4].y, wl[q4].z, wl[q4].w};
            union { v8h v; v2h h2[4]; } ou;
#pragma unroll
            for (int e = 0; e < 4; ++e) {
                unsigned v = (unsigned)vals[e];
                // K=2e from high nibble (low half), K=2e+1 from low nibble (high half)
                unsigned t = (((v >> 4) & 0xFu) | ((v & 0xFu) << 16)) | 0x64006400u;
                union { unsigned u; v2h h; } uh;
                uh.u = t;
                ou.h2[e] = uh.h * s2 + c2;     // v_pk_fma_f16
            }
            *(v8h*)(&Bs[buf][wr * LDT + wh * 32 + q4 * 8]) = ou.v;
        }
    };

    // WMMA compute on buffer p
    auto compute = [&](int p) {
        const _Float16* Ab = As[p];
        const _Float16* Bb = Bs[p];
#pragma unroll
        for (int kk = 0; kk < BK; kk += 32) {
            v16h afr[4];
#pragma unroll
            for (int i = 0; i < 4; ++i) {
                int row = wm * 64 + i * 16 + lane16;
                union { v16h v; v8h h[2]; } u;
                u.h[0] = *(const v8h*)(Ab + row * LDT + kk + sel * 8);
                u.h[1] = *(const v8h*)(Ab + row * LDT + kk + 16 + sel * 8);
                afr[i] = u.v;
            }
            v16h bfr[2];
#pragma unroll
            for (int j = 0; j < 2; ++j) {
                int col = wn * 32 + j * 16 + lane16;
                const _Float16* pp = Bb + col * LDT + kk + sel * 8;
                union { v16h v; v8h h[2]; } u;
                u.h[0] = *(const v8h*)(pp);
                u.h[1] = *(const v8h*)(pp + 16);
                bfr[j] = u.v;
            }
#pragma unroll
            for (int i = 0; i < 4; ++i)
#pragma unroll
                for (int j = 0; j < 2; ++j)
                    acc[i][j] = __builtin_amdgcn_wmma_f32_16x16x32_f16(
                        false, afr[i], false, bfr[j],
                        (short)0, acc[i][j], false, false);
        }
    };

    // ---- software pipeline: prologue fills buffer 0 ----
    issue_x(0, 0);
    load_w(0);
    dequant_store(0);
    asm volatile("s_wait_asynccnt 0" ::: "memory");
    __syncthreads();

    for (int kb = 0; kb < I_DIM; kb += BK) {
        const int p = (kb / BK) & 1;
        const bool has_next = (kb + BK) < I_DIM;

        if (has_next) {
            issue_x(1 - p, kb + BK);        // async: overlaps with WMMA below
            load_w(kb + BK);                // global loads overlap with WMMA
        }

        compute(p);

        if (has_next)
            dequant_store(1 - p);           // pk-fma co-executes around WMMA tail

        asm volatile("s_wait_asynccnt 0" ::: "memory");
        __syncthreads();
    }

    // ---- epilogue: add bias, store fp16 ----
#pragma unroll
    for (int j = 0; j < 2; ++j) {
        int n = bn0 + wn * 32 + j * 16 + lane16;
        float bv = (float)bias[n];
#pragma unroll
        for (int i = 0; i < 4; ++i) {
            int mbase = bm0 + wm * 64 + i * 16 + sel * 8;
#pragma unroll
            for (int e = 0; e < 8; ++e) {
                y[(size_t)(mbase + e) * O_DIM + n] = (_Float16)(acc[i][j][e] + bv);
            }
        }
    }
}

extern "C" void kernel_launch(void* const* d_in, const int* in_sizes, int n_in,
                              void* d_out, int out_size, void* d_ws, size_t ws_size,
                              hipStream_t stream) {
    const _Float16* x    = (const _Float16*)d_in[0];
    const int*      wq   = (const int*)d_in[1];
    const _Float16* zs   = (const _Float16*)d_in[2];
    const _Float16* bias = (const _Float16*)d_in[3];
    _Float16*       y    = (_Float16*)d_out;

    dim3 grid(O_DIM / BN, M_TOT / BM);   // (86, 64)
    w4a16_wmma_kernel<<<grid, dim3(256), 0, stream>>>(x, wq, zs, bias, y);
}
